// Self_Attention_28372553957592
// MI455X (gfx1250) — matrix operand order repaired and checked by reading, not compiled, not run on hardware
//
#include <hip/hip_runtime.h>
#include <hip/hip_bf16.h>
#include <stdint.h>

// ---------- CDNA5 WMMA vector types ----------
typedef __attribute__((ext_vector_type(16))) __bf16 v16bf;
typedef __attribute__((ext_vector_type(8)))  float  v8f;

// Native bf16 converts (hardware v_cvt, instead of manual integer rounding)
static __device__ __forceinline__ unsigned short f2bf_bits(float f) {
    __bf16 b = (__bf16)f;
    unsigned short u;
    __builtin_memcpy(&u, &b, 2);
    return u;
}
static __device__ __forceinline__ float bf2f(unsigned short h) {
    __bf16 b;
    __builtin_memcpy(&b, &h, 2);
    return (float)b;
}

// ---- CDNA5 async global->LDS copy (ASYNCcnt-tracked, no VGPR round trip) ----
static __device__ __forceinline__ void async_b128(unsigned short* lds, const void* g) {
    // low 32 bits of a generic LDS pointer are the LDS byte address (ISA 10.2)
    unsigned loff = (unsigned)(uintptr_t)lds;
    asm volatile("global_load_async_to_lds_b128 %0, %1, off"
                 :: "v"(loff), "v"(g) : "memory");
}
static __device__ __forceinline__ void wait_async0() {
    asm volatile("s_wait_asynccnt 0x0" ::: "memory");
}

// raw staging registers for f32 (or bf16-transpose) tiles
struct Raw16 { union { float4 f[4]; uint4 q[2]; unsigned short h[16]; }; };

union Frag { v16bf v; uint4 q[2]; };

// ---------------------------------------------------------------------------
// Tiled WMMA GEMM:  C[M][N] = scale * (A[M][K] @ op(B))  (+ causal mask) (+ bias)
//   BTRANS=true : B is [N][K] (torch Linear weight, y = x @ W^T)
//   BTRANS=false: B is [K][N] (attention V)
// f32 inputs are converted to bf16 via native cvt on the way into LDS;
// bf16 inputs stream straight into LDS with global_load_async_to_lds_b128.
// Double-buffered LDS, one barrier per K-step; 8 wmma per K-step per wave.
// Block: 256 threads (8 wave32), 128x128 tile, K-step 32.
// ---------------------------------------------------------------------------
template <typename TA, typename TB, bool BTRANS, bool CAUSAL, bool BIAS, bool OUT_BF16>
__launch_bounds__(256)
__global__ void gemm_wmma(const TA* __restrict__ A, const TB* __restrict__ Bm,
                          void* __restrict__ Cv, const float* __restrict__ bias,
                          int M, int N, int Kd,
                          long long sA, long long sB, long long sC, float scale)
{
    constexpr bool A_BF16    = (sizeof(TA) == 2);
    constexpr bool B_BF16    = (sizeof(TB) == 2);
    constexpr bool USE_ASYNC = A_BF16 || (B_BF16 && !BTRANS);

    __shared__ __align__(16) unsigned short As[2][128 * 32];   // [m][k] bf16, 2x8KB
    __shared__ __align__(16) unsigned short Bs[2][32 * 128];   // [k][n] bf16, 2x8KB

    const long long z = blockIdx.z;
    A  += z * sA;
    Bm += z * sB;
    unsigned short* C16 = reinterpret_cast<unsigned short*>(Cv) + z * sC;
    float*          C32 = reinterpret_cast<float*>(Cv)          + z * sC;

    const int n0   = blockIdx.x * 128;
    const int m0   = blockIdx.y * 128;
    const int t    = threadIdx.x;
    const int lane = t & 31;
    const int wave = t >> 5;
    const int wm   = wave & 1;    // row half (64 rows)
    const int wn   = wave >> 1;   // col quarter (32 cols)

    // staging coordinates: 16 elements per thread per tile
    const int arow = t >> 1, acb = (t & 1) << 4;        // A: 2 thr/row, 32 k
    const int bnl  = t >> 1, bcb = (t & 1) << 4;        // B trans: 2 thr/row
    const int bkl  = t & 31, bnb = (t >> 5) << 4;       // B direct: 8 thr/k-row

    const TA* aG0 = A + (long long)(m0 + arow) * Kd + acb;
    const TB* bG0 = BTRANS ? (Bm + (long long)(n0 + bnl) * Kd + bcb)
                           : (Bm + (long long)bkl * N + n0 + bnb);
    const long long bStep = BTRANS ? 32 : (long long)32 * N;

    Raw16 ra, rb;

    // ---- issue global traffic for tile kt into buffer `buf` ----
    auto issue = [&](int kt, int buf) {
        const TA* ag = aG0 + (long long)kt * 32;
        unsigned short* al = &As[buf][arow * 32 + acb];
        if (A_BF16) {
            async_b128(al,     (const void*)ag);
            async_b128(al + 8, (const void*)((const unsigned short*)(const void*)ag + 8));
        } else {
            const float4* p = (const float4*)(const void*)ag;
            ra.f[0] = p[0]; ra.f[1] = p[1]; ra.f[2] = p[2]; ra.f[3] = p[3];
        }
        const TB* bg = bG0 + bStep * kt;
        if (BTRANS) {
            if (B_BF16) {
                const uint4* p = (const uint4*)(const void*)bg;
                rb.q[0] = p[0]; rb.q[1] = p[1];
            } else {
                const float4* p = (const float4*)(const void*)bg;
                rb.f[0] = p[0]; rb.f[1] = p[1]; rb.f[2] = p[2]; rb.f[3] = p[3];
            }
        } else {
            unsigned short* bl = &Bs[buf][bkl * 128 + bnb];
            if (B_BF16) {
                async_b128(bl,     (const void*)bg);
                async_b128(bl + 8, (const void*)((const unsigned short*)(const void*)bg + 8));
            } else {
                const float4* p = (const float4*)(const void*)bg;
                rb.f[0] = p[0]; rb.f[1] = p[1]; rb.f[2] = p[2]; rb.f[3] = p[3];
            }
        }
    };
    // ---- convert/scatter staged registers into buffer `buf` ----
    auto commit = [&](int buf) {
        if (!A_BF16) {
            unsigned short tmp[16];
            const float* f = (const float*)&ra;
#pragma unroll
            for (int j = 0; j < 16; ++j) tmp[j] = f2bf_bits(f[j]);
            uint4* dst = (uint4*)&As[buf][arow * 32 + acb];
            dst[0] = ((const uint4*)tmp)[0]; dst[1] = ((const uint4*)tmp)[1];
        }
        if (BTRANS) {
#pragma unroll
            for (int j = 0; j < 16; ++j) {
                unsigned short hv = B_BF16 ? rb.h[j] : f2bf_bits(((const float*)&rb)[j]);
                Bs[buf][(bcb + j) * 128 + bnl] = hv;
            }
        } else if (!B_BF16) {
            unsigned short tmp[16];
            const float* f = (const float*)&rb;
#pragma unroll
            for (int j = 0; j < 16; ++j) tmp[j] = f2bf_bits(f[j]);
            uint4* dst = (uint4*)&Bs[buf][bkl * 128 + bnb];
            dst[0] = ((const uint4*)tmp)[0]; dst[1] = ((const uint4*)tmp)[1];
        }
    };

    v8f c[4][2];
#pragma unroll
    for (int i = 0; i < 4; ++i)
#pragma unroll
        for (int j = 0; j < 2; ++j) {
            v8f zv = {0.f, 0.f, 0.f, 0.f, 0.f, 0.f, 0.f, 0.f};
            c[i][j] = zv;
        }

    // ---- prologue: stage tile 0 ----
    issue(0, 0);
    commit(0);
    if (USE_ASYNC) wait_async0();
    __syncthreads();

    const int nK = Kd >> 5;
    for (int kt = 0; kt < nK; ++kt) {
        const int  cur  = kt & 1, nxt = cur ^ 1;
        const bool more = (kt + 1) < nK;
        if (more) issue(kt + 1, nxt);                    // overlap with compute below
        if (kt + 2 < nK)
            __builtin_prefetch(aG0 + (long long)(kt + 2) * 32, 0, 1);

        // ---- fragments per ISA 7.12.2 layouts ----
        Frag a[4], b[2];
        const int off = (lane >> 4) << 3;   // lanes 0-15: k {0..7,16..23}; 16-31: {8..15,24..31}
#pragma unroll
        for (int i = 0; i < 4; ++i) {
            const int row = wm * 64 + i * 16 + (lane & 15);
            a[i].q[0] = *reinterpret_cast<const uint4*>(&As[cur][row * 32 + off]);
            a[i].q[1] = *reinterpret_cast<const uint4*>(&As[cur][row * 32 + off + 16]);
        }
#pragma unroll
        for (int j = 0; j < 2; ++j) {
            const int col = wn * 32 + j * 16 + (lane & 15);   // lane = n column, 16 halves = 16 k values (same k order as A)
            unsigned short tb[16];
#pragma unroll
            for (int s = 0; s < 16; ++s) tb[s] = Bs[cur][(off + (s < 8 ? s : s + 8)) * 128 + col];
            __builtin_memcpy(&b[j], tb, 32);
        }
#pragma unroll
        for (int i = 0; i < 4; ++i)
#pragma unroll
            for (int j = 0; j < 2; ++j)
                c[i][j] = __builtin_amdgcn_wmma_f32_16x16x32_bf16(
                    false, a[i].v, false, b[j].v, (short)0, c[i][j], false, false);

        if (more) commit(nxt);
        if (USE_ASYNC) wait_async0();
        __syncthreads();
    }

    // ---- epilogue: C layout = VGPR r, lanes 0-15: M=r,N=lane; lanes 16-31: M=8+r ----
#pragma unroll
    for (int i = 0; i < 4; ++i) {
#pragma unroll
        for (int j = 0; j < 2; ++j) {
            const int gn = n0 + wn * 32 + j * 16 + (lane & 15);
#pragma unroll
            for (int r = 0; r < 8; ++r) {
                const int gm = m0 + wm * 64 + i * 16 + r + ((lane >> 4) << 3);
                float val = c[i][j][r] * scale;
                if (CAUSAL) { if (gn > gm) val = -1e30f; }   // within-batch indices
                if (BIAS)   val += bias[gn];
                if (OUT_BF16) C16[(long long)gm * N + gn] = f2bf_bits(val);
                else          C32[(long long)gm * N + gn] = val;
            }
        }
    }
}

// ---------------------------------------------------------------------------
// Row softmax on bf16 scores, in place.  One block (256 threads) per row of
// 2048 columns; f32 math; masked entries (-1e30) underflow to zero.
// ---------------------------------------------------------------------------
__launch_bounds__(256)
__global__ void softmax_rows(unsigned short* __restrict__ P, int cols)
{
    __shared__ float red[256];
    const int t = threadIdx.x;
    unsigned short* p = P + (long long)blockIdx.x * cols;   // cols == 2048

    float v[8];
    float mx = -1e30f;
#pragma unroll
    for (int i = 0; i < 8; ++i) { v[i] = bf2f(p[t + (i << 8)]); mx = fmaxf(mx, v[i]); }
    red[t] = mx; __syncthreads();
    for (int s = 128; s > 0; s >>= 1) { if (t < s) red[t] = fmaxf(red[t], red[t + s]); __syncthreads(); }
    mx = red[0]; __syncthreads();

    float sum = 0.f;
#pragma unroll
    for (int i = 0; i < 8; ++i) { v[i] = __expf(v[i] - mx); sum += v[i]; }
    red[t] = sum; __syncthreads();
    for (int s = 128; s > 0; s >>= 1) { if (t < s) red[t] += red[t + s]; __syncthreads(); }
    const float inv = 1.0f / red[0];   // >= 1 (diagonal term)

#pragma unroll
    for (int i = 0; i < 8; ++i) p[t + (i << 8)] = f2bf_bits(v[i] * inv);
}

// ---------------------------------------------------------------------------
extern "C" void kernel_launch(void* const* d_in, const int* in_sizes, int n_in,
                              void* d_out, int out_size, void* d_ws, size_t ws_size,
                              hipStream_t stream)
{
    (void)in_sizes; (void)n_in; (void)out_size; (void)ws_size;

    const float* x  = (const float*)d_in[0];   // [B,S,D] f32
    const float* Wq = (const float*)d_in[1];   // [D,D]
    const float* Wk = (const float*)d_in[2];
    const float* Wv = (const float*)d_in[3];
    const float* Wo = (const float*)d_in[4];
    const float* bo = (const float*)d_in[5];   // [D]

    constexpr int Bn = 4, S = 2048, D = 1024;
    constexpr int M  = Bn * S;                 // 8192
    const float inv_sqrt_d = 0.03125f;         // 1/sqrt(1024)

    // workspace layout (bf16 stored as ushort):
    //   Qb,Kb,Vb,Ob : [M][D]  (16.78 MB each)
    //   P           : [Bn][S][S] (33.55 MB)        total ~100.7 MB
    const size_t E = (size_t)M * D;
    unsigned short* Qb = (unsigned short*)d_ws;
    unsigned short* Kb = Qb + E;
    unsigned short* Vb = Kb + E;
    unsigned short* Ob = Vb + E;
    unsigned short* P  = Ob + E;

    const dim3 blk(256);
    const long long sQK = (long long)S * D;    // per-batch stride of Q/K/V/O
    const long long sP  = (long long)S * S;    // per-batch stride of scores

    // 1) Q,K,V projections: [M,D] = x[M,D] @ W[D,D]^T  -> bf16
    gemm_wmma<float, float, true, false, false, true>
        <<<dim3(D / 128, M / 128, 1), blk, 0, stream>>>(x, Wq, Qb, nullptr, M, D, D, 0, 0, 0, 1.0f);
    gemm_wmma<float, float, true, false, false, true>
        <<<dim3(D / 128, M / 128, 1), blk, 0, stream>>>(x, Wk, Kb, nullptr, M, D, D, 0, 0, 0, 1.0f);
    gemm_wmma<float, float, true, false, false, true>
        <<<dim3(D / 128, M / 128, 1), blk, 0, stream>>>(x, Wv, Vb, nullptr, M, D, D, 0, 0, 0, 1.0f);

    // 2) Scores: per batch, P = causal(Q @ K^T / sqrt(D)) -> bf16  (A async-to-LDS)
    gemm_wmma<unsigned short, unsigned short, true, true, false, true>
        <<<dim3(S / 128, S / 128, Bn), blk, 0, stream>>>(Qb, Kb, P, nullptr, S, S, D, sQK, sQK, sP, inv_sqrt_d);

    // 3) Row softmax in place
    softmax_rows<<<dim3(Bn * S), blk, 0, stream>>>(P, S);

    // 4) O = P @ V  (B is [K][N]; both operands async-to-LDS) -> bf16
    gemm_wmma<unsigned short, unsigned short, false, false, false, true>
        <<<dim3(D / 128, S / 128, Bn), blk, 0, stream>>>(P, Vb, Ob, nullptr, S, D, S, sP, sQK, sQK, 1.0f);

    // 5) out = O @ Wo^T + bo -> f32  (A async-to-LDS)
    gemm_wmma<unsigned short, float, true, false, true, false>
        <<<dim3(D / 128, M / 128, 1), blk, 0, stream>>>(Ob, Wo, d_out, bo, M, D, D, 0, 0, 0, 1.0f);
}
